// spiking_transformer_3848290697755
// MI455X (gfx1250) — matrix operand order repaired and checked by reading, not compiled, not run on hardware
//
#include <hip/hip_runtime.h>
#include <hip/hip_bf16.h>

typedef _Float16 h16;
typedef __attribute__((ext_vector_type(16))) _Float16 v16h;
typedef __attribute__((ext_vector_type(8)))  _Float16 v8h;
typedef __attribute__((ext_vector_type(8)))  float    v8f;
typedef __attribute__((ext_vector_type(4)))  float    v4f;

#define TT   4
#define BB   16
#define NN   1024
#define CC   256
#define CHH  1024
#define LTAU 0.5f

// ---------------------------------------------------------------------------
// fp32 -> fp16 weight conversion
// ---------------------------------------------------------------------------
__global__ void cvt_f32_f16_kernel(const float* __restrict__ src,
                                   h16* __restrict__ dst, int n) {
  int i = blockIdx.x * blockDim.x + threadIdx.x;
  if (i < n) dst[i] = (h16)src[i];
}

// ---------------------------------------------------------------------------
// LIF over T on the raw input x [T][B][C][N] fp32 -> spikes xs [T][B][N][C] f16
// One thread per (b,n,c); c fastest -> contiguous fp16 stores; strided fp32
// reads are L2-resident (x = 67MB < 192MB L2).
// ---------------------------------------------------------------------------
__global__ void lif_input_kernel(const float* __restrict__ x,
                                 h16* __restrict__ xs) {
  int tid = blockIdx.x * blockDim.x + threadIdx.x;   // B*N*C threads
  int c = tid & (CC - 1);
  int n = (tid >> 8) & (NN - 1);
  int b = tid >> 18;
  float mem = 0.f;
  #pragma unroll
  for (int t = 0; t < TT; ++t) {
    float xv = x[(size_t)((t * BB + b) * CC + c) * NN + n];
    mem = mem * LTAU + xv;
    float s = (mem >= 1.0f) ? 1.f : 0.f;
    xs[(size_t)((t * BB + b) * NN + n) * CC + c] = (h16)s;
    mem *= (1.f - s);
  }
}

// ---------------------------------------------------------------------------
// Attention combine: one wave per (b, head, n); lane = d (0..31).
//   mems = EMA over t of q; qsum = sum_d(mems + q); attn = lif(qsum, thr=0.5);
//   y = attn * k.  q,k,y stored [t][b][n][c] fp16, c = head*32 + lane.
// ---------------------------------------------------------------------------
__global__ void attn_kernel(const h16* __restrict__ q,
                            const h16* __restrict__ k,
                            const float* __restrict__ alpha_p,
                            h16* __restrict__ y) {
  int flat = blockIdx.x * blockDim.x + threadIdx.x;
  int wid  = flat >> 5;
  int lane = threadIdx.x & 31;
  int n  = wid & (NN - 1);
  int hh = (wid >> 10) & 7;
  int b  = wid >> 13;
  float alpha = alpha_p[0];
  int c = hh * 32 + lane;

  float a_mem = 0.f, mprev = 0.f, qprev = 0.f;
  #pragma unroll
  for (int t = 0; t < TT; ++t) {
    size_t base = (size_t)((t * BB + b) * NN + n) * CC + c;
    float qv = (float)q[base];
    float kv = (float)k[base];
    float mcur = (t == 0) ? qv : (alpha * mprev + (1.f - alpha) * qprev);
    float contrib = mcur + qv;
    // wave32 head-wise reduction over d=32 lanes
    #pragma unroll
    for (int off = 16; off > 0; off >>= 1)
      contrib += __shfl_xor(contrib, off, 32);
    a_mem = a_mem * LTAU + contrib;
    float s = (a_mem >= 0.5f) ? 1.f : 0.f;
    a_mem *= (1.f - s);
    y[base] = (h16)(s * kv);
    mprev = mcur; qprev = qv;
  }
}

// ---------------------------------------------------------------------------
// Fused WMMA GEMM + BN + epilogue, 32x32 output tile per wave (2x2 fragments).
//   D[o,n] = sum_c W[o,c] * S[t,b,n,c]   (S = fp16 spikes, [t][b][n][CIN])
//   val    = g[o]*(D + b1[o]) + b0[o]  ==  g[o]*D + bb[o],  bb = g*b1 + b0
// Per k-step: 2 A-fragment loads + 2 B-fragment loads feed 4 WMMAs.
// The wave loops t=0..3 internally so the LIF membrane (8 floats per lane per
// fragment) lives in registers across time steps, and weight fragments get 4x
// temporal reuse through L1. Only gv/bb are hoisted (32 floats/lane) to keep
// VGPR pressure below the spill threshold; epilogue is one FMA per element.
// Fragment layouts used here (16-bit A 16x32 / B 32x16 / f32 C 16x16):
//   A: lane<16 K 0..15, lane>=16 K 16..31, M = lane%16 (two 16B loads)
//   B: lane<16 K 0..15, lane>=16 K 16..31, N = lane%16 (one 32B load)
//   C/D: VGPR r <-> M = r + 8*(lane>=16), N = lane%16 -> 8 consecutive output
//        channels per lane in the [n][o] layout -> single b128 spike store.
// MODE 0: store LIF spikes (q / k / mlp-hidden)
// MODE 1: proj: xr = x(orig,[c][n]) + val -> store xr f32 [n][o] + LIF spikes
// MODE 2: final: out = xr([n][o]) + val -> store f32 to d_out [t][b][o][n]
// ---------------------------------------------------------------------------
template <int CIN, int COUT, int MODE, bool HAS_B1>
__global__ void gemm_lif_kernel(const h16*  __restrict__ Wh,     // [COUT][CIN] f16
                                const h16*  __restrict__ S,      // [T*B][NN][CIN] f16
                                const float* __restrict__ g,
                                const float* __restrict__ b1,
                                const float* __restrict__ b0,
                                h16*  __restrict__ out_spk,      // [T*B][NN][COUT] f16
                                float* __restrict__ out_f32,
                                const float* __restrict__ xres) {
  constexpr int NT = NN / 32;      // 32  n-supertiles
  constexpr int OT = COUT / 32;    // o-supertiles
  int flat = blockIdx.x * blockDim.x + threadIdx.x;
  int wid  = flat >> 5;
  int lane = threadIdx.x & 31;
  int nt  = wid % NT;
  int rem = wid / NT;
  int ot  = rem % OT;
  int b   = rem / OT;

  int lane16 = lane & 15;
  bool hi    = lane >= 16;
  int mbase  = hi ? 8 : 0;

  int obase[2], ncol[2];
  const h16* arow[2];
  #pragma unroll
  for (int mi = 0; mi < 2; ++mi) {
    obase[mi] = ot * 32 + mi * 16 + mbase;
    arow[mi]  = Wh + (size_t)(ot * 32 + mi * 16 + lane16) * CIN + (hi ? 16 : 0);
  }
  #pragma unroll
  for (int ni = 0; ni < 2; ++ni) ncol[ni] = nt * 32 + ni * 16 + lane16;

  // hoist fused affine params: val = gv*acc + bb  (bb = g*b1 + b0)
  float gv[2][8], bb[2][8];
  #pragma unroll
  for (int mi = 0; mi < 2; ++mi)
    #pragma unroll
    for (int r = 0; r < 8; ++r) {
      int o = obase[mi] + r;
      float gg = g[o];
      gv[mi][r] = gg;
      bb[mi][r] = HAS_B1 ? fmaf(gg, b1[o], b0[o]) : b0[o];
    }

  float mem[2][2][8];
  #pragma unroll
  for (int mi = 0; mi < 2; ++mi)
    #pragma unroll
    for (int ni = 0; ni < 2; ++ni)
      #pragma unroll
      for (int r = 0; r < 8; ++r) mem[mi][ni][r] = 0.f;

  for (int t = 0; t < TT; ++t) {
    v8f acc[2][2];
    #pragma unroll
    for (int mi = 0; mi < 2; ++mi)
      #pragma unroll
      for (int ni = 0; ni < 2; ++ni) acc[mi][ni] = {};

    const h16* srow[2];
    #pragma unroll
    for (int ni = 0; ni < 2; ++ni)
      srow[ni] = S + ((size_t)(t * BB + b) * NN + ncol[ni]) * CIN + (hi ? 16 : 0);

    #pragma unroll 2
    for (int kk = 0; kk < CIN / 32; ++kk) {
      v16h af[2], bf[2];
      #pragma unroll
      for (int mi = 0; mi < 2; ++mi) {
        v8h a0 = *(const v8h*)(arow[mi] + kk * 32);
        v8h a1 = *(const v8h*)(arow[mi] + kk * 32 + 8);
        af[mi] = __builtin_shufflevector(a0, a1, 0, 1, 2, 3, 4, 5, 6, 7,
                                                 8, 9, 10, 11, 12, 13, 14, 15);
      }
      #pragma unroll
      for (int ni = 0; ni < 2; ++ni)
        bf[ni] = *(const v16h*)(srow[ni] + kk * 32);
      #pragma unroll
      for (int mi = 0; mi < 2; ++mi)
        #pragma unroll
        for (int ni = 0; ni < 2; ++ni)
          acc[mi][ni] = __builtin_amdgcn_wmma_f32_16x16x32_f16(
              false, af[mi], false, bf[ni], (short)0, acc[mi][ni], false, false);
    }

    // ---- epilogue: fused BN affine + LIF/residual per 16x16 fragment ----
    #pragma unroll
    for (int mi = 0; mi < 2; ++mi) {
      #pragma unroll
      for (int ni = 0; ni < 2; ++ni) {
        size_t outb = ((size_t)(t * BB + b) * NN + ncol[ni]) * COUT + obase[mi];
        if (MODE == 0) {
          v8h spk;
          #pragma unroll
          for (int r = 0; r < 8; ++r) {
            float val = fmaf(gv[mi][r], acc[mi][ni][r], bb[mi][r]);
            float& m = mem[mi][ni][r];
            m = m * LTAU + val;
            float s = (m >= 1.0f) ? 1.f : 0.f;
            m *= (1.f - s);
            spk[r] = (h16)s;
          }
          *(v8h*)(out_spk + outb) = spk;
        } else if (MODE == 1) {
          v8h spk;
          float xrv[8];
          #pragma unroll
          for (int r = 0; r < 8; ++r) {
            int o = obase[mi] + r;
            float val = fmaf(gv[mi][r], acc[mi][ni][r], bb[mi][r]);
            float xr = xres[((size_t)(t * BB + b) * COUT + o) * NN + ncol[ni]] + val;
            xrv[r] = xr;
            float& m = mem[mi][ni][r];
            m = m * LTAU + xr;
            float s = (m >= 1.0f) ? 1.f : 0.f;
            m *= (1.f - s);
            spk[r] = (h16)s;
          }
          v4f lo  = {xrv[0], xrv[1], xrv[2], xrv[3]};
          v4f hi4 = {xrv[4], xrv[5], xrv[6], xrv[7]};
          *(v4f*)(out_f32 + outb)     = lo;
          *(v4f*)(out_f32 + outb + 4) = hi4;
          *(v8h*)(out_spk + outb) = spk;
        } else { // MODE 2: final residual, scatter to [t][b][o][n]
          #pragma unroll
          for (int r = 0; r < 8; ++r) {
            int o = obase[mi] + r;
            float val = fmaf(gv[mi][r], acc[mi][ni][r], bb[mi][r]);
            float res = xres[outb + r] + val;
            out_f32[((size_t)(t * BB + b) * COUT + o) * NN + ncol[ni]] = res;
          }
        }
      }
    }
  }
}

// ---------------------------------------------------------------------------
extern "C" void kernel_launch(void* const* d_in, const int* in_sizes, int n_in,
                              void* d_out, int out_size, void* d_ws, size_t ws_size,
                              hipStream_t stream) {
  const float* x         = (const float*)d_in[0];
  const float* q_w       = (const float*)d_in[1];
  const float* q_g       = (const float*)d_in[2];
  const float* q_b       = (const float*)d_in[3];
  const float* k_w       = (const float*)d_in[4];
  const float* k_g       = (const float*)d_in[5];
  const float* k_b       = (const float*)d_in[6];
  const float* proj_w    = (const float*)d_in[7];
  const float* proj_bias = (const float*)d_in[8];
  const float* proj_g    = (const float*)d_in[9];
  const float* proj_b2   = (const float*)d_in[10];
  const float* mem_alpha = (const float*)d_in[11];
  const float* w1        = (const float*)d_in[12];
  const float* b1        = (const float*)d_in[13];
  const float* bn1_g     = (const float*)d_in[14];
  const float* bn1_b     = (const float*)d_in[15];
  const float* w2        = (const float*)d_in[16];
  const float* b2        = (const float*)d_in[17];
  const float* bn2_g     = (const float*)d_in[18];
  const float* bn2_b     = (const float*)d_in[19];
  float* out = (float*)d_out;

  // ---- workspace layout (bytes) ----
  // xr  [T][B][N][C]  f32 : 67,108,864
  // xs  [T][B][N][C]  f16 : 33,554,432   (reused as zs after proj stage)
  // big region: q, k, y (f16, 33.5MB each) -> overlaid by h1 (134.2MB) after K4
  // fp16 weights at tail
  char* ws = (char*)d_ws;
  float* xr = (float*)ws;
  h16*   xs = (h16*)(ws + 67108864);
  char*  big = ws + 100663296;
  h16*   q  = (h16*)big;
  h16*   k  = (h16*)(big + 33554432);
  h16*   y  = (h16*)(big + 67108864);
  h16*   h1 = (h16*)big;                       // overlays q/k/y (dead by then)
  h16*   wts  = (h16*)(ws + 234881024);
  h16*   qw_h = wts;
  h16*   kw_h = wts + 65536;
  h16*   pw_h = wts + 131072;
  h16*   w1_h = wts + 196608;
  h16*   w2_h = wts + 458752;

  // ---- weight fp32 -> fp16 ----
  cvt_f32_f16_kernel<<<256, 256, 0, stream>>>(q_w, qw_h, 65536);
  cvt_f32_f16_kernel<<<256, 256, 0, stream>>>(k_w, kw_h, 65536);
  cvt_f32_f16_kernel<<<256, 256, 0, stream>>>(proj_w, pw_h, 65536);
  cvt_f32_f16_kernel<<<1024, 256, 0, stream>>>(w1, w1_h, 262144);
  cvt_f32_f16_kernel<<<1024, 256, 0, stream>>>(w2, w2_h, 262144);

  // ---- stage 1: xs = lif(x), transposed to [t][b][n][c] f16 ----
  lif_input_kernel<<<16384, 256, 0, stream>>>(x, xs);

  // ---- stage 2: q = lif(bn(q_w @ xs)), k likewise (WMMA, fused LIF) ----
  // waves = B * (COUT/32) * (NN/32) = 16*8*32 = 4096 -> 512 blocks of 8 waves
  gemm_lif_kernel<CC, CC, 0, false><<<512, 256, 0, stream>>>(
      qw_h, xs, q_g, nullptr, q_b, q, nullptr, nullptr);
  gemm_lif_kernel<CC, CC, 0, false><<<512, 256, 0, stream>>>(
      kw_h, xs, k_g, nullptr, k_b, k, nullptr, nullptr);

  // ---- stage 3: attention combine -> y ----
  attn_kernel<<<16384, 256, 0, stream>>>(q, k, mem_alpha, y);

  // ---- stage 4: xr = x + bn(proj_w @ y + proj_bias); zs = lif(xr) (fused) ----
  gemm_lif_kernel<CC, CC, 1, true><<<512, 256, 0, stream>>>(
      pw_h, y, proj_g, proj_bias, proj_b2, /*zs=*/xs, /*xr=*/xr, /*orig x=*/x);

  // ---- stage 5: h1 = lif(bn(w1 @ zs + b1)) ----
  // waves = 16*32*32 = 16384 -> 2048 blocks
  gemm_lif_kernel<CC, CHH, 0, true><<<2048, 256, 0, stream>>>(
      w1_h, xs, bn1_g, b1, bn1_b, h1, nullptr, nullptr);

  // ---- stage 6: out = xr + bn(w2 @ h1 + b2), scattered to [t][b][c][h][w] ----
  gemm_lif_kernel<CHH, CC, 2, true><<<512, 256, 0, stream>>>(
      w2_h, h1, bn2_g, b2, bn2_b, nullptr, out, xr);
}
